// PseudoSPNet_78305843741425
// MI455X (gfx1250) — matrix operand order repaired and checked by reading, not compiled, not run on hardware
//
#include <hip/hip_runtime.h>

typedef __attribute__((ext_vector_type(16))) _Float16 v16h;
typedef __attribute__((ext_vector_type(8)))  _Float16 v8h;
typedef __attribute__((ext_vector_type(8)))  float    v8f;
typedef __attribute__((ext_vector_type(4)))  float    f4;

__device__ __forceinline__ float lrelu(float v){ return v >= 0.f ? v : 0.2f*v; }

// ---------------- WMMA fragment helpers (CDNA5 16x16x32 f16 layouts) --------
// A (16x32, MxK) row-major source [row][128]: lane l: M=l%15.. M=l&15, half=l>>4.
// VGPR0..3 hold K = half*8 + {0..7}; VGPR4..7 hold K = 16 + half*8 + {0..7}.
__device__ __forceinline__ v16h load_a_frag(const _Float16* __restrict__ A, int kc){
  int lane = threadIdx.x & 31;
  int row  = lane & 15;
  int hf   = lane >> 4;
  const _Float16* p = A + row*128 + kc + hf*8;
  v8h lo = *(const v8h*)(p);
  v8h hi = *(const v8h*)(p + 16);
  v16h a;
#pragma unroll
  for (int i = 0; i < 8; ++i){ a[i] = lo[i]; a[8+i] = hi[i]; }
  return a;
}

// B (32x16, KxN) from row-major weights W[n][128] (B[k][n] = W[n][k]):
// lane l: N = l&15, khalf = l>>4; e<8 -> K = kc + khalf*8 + e; e>=8 -> K = kc + 16 + khalf*8 + (e-8).
__device__ __forceinline__ v16h load_b_frag(const _Float16* __restrict__ W, int n0, int kc){
  int lane = threadIdx.x & 31;
  int n  = lane & 15;
  int kh = lane >> 4;
  const _Float16* p = W + (size_t)(n0 + n)*128 + kc + kh*8;
  v8h lo = *(const v8h*)(p);
  v8h hi = *(const v8h*)(p + 16);
  v16h b;
#pragma unroll
  for (int i = 0; i < 8; ++i){ b[i] = lo[i]; b[8+i] = hi[i]; }
  return b;
}

__device__ __forceinline__ v8f wmma_f16(v16h a, v16h b, v8f c){
  return __builtin_amdgcn_wmma_f32_16x16x32_f16(false, a, false, b, (short)0, c, false, false);
}

// ---------------- K0: zero stats + convert weights to f16 -------------------
__global__ void init_convert(const float* __restrict__ sw3, const float* __restrict__ cw3,
                             _Float16* __restrict__ sw3h, _Float16* __restrict__ cw3eh,
                             float* __restrict__ stats){
  int idx = blockIdx.x*256 + threadIdx.x;            // grid 512 -> 131072
  if (idx < 131072) sw3h[idx] = (_Float16)sw3[idx];  // sw3: (1024,128)
  if (idx < 65536){                                  // even rows of cw3 -> (512,128)
    int j = idx >> 7, k = idx & 127;
    cw3eh[idx] = (_Float16)cw3[(size_t)(2*j)*128 + k];
  }
  if (idx < 1408) stats[idx] = 0.f;                  // all BN accumulators
}

// ---------------- STN conv1+conv2 (3->64->128), f16 output ------------------
__global__ void stn_conv12(const float* __restrict__ x,
                           const float* __restrict__ sw1, const float* __restrict__ sb1,
                           const float* __restrict__ sw2, const float* __restrict__ sb2,
                           _Float16* __restrict__ h2f16){
  __shared__ float W2[128*64];
  __shared__ float W1[192], B1[64], B2[128];
  int tid = threadIdx.x;
  for (int i = tid; i < 8192; i += 256) W2[i] = sw2[i];
  if (tid < 192) W1[tid] = sw1[tid];
  if (tid < 64)  B1[tid] = sb1[tid];
  if (tid < 128) B2[tid] = sb2[tid];
  __syncthreads();
  int p = blockIdx.x*256 + tid;                      // 16384 points
  int b = p >> 10, n = p & 1023;
  float x0 = x[b*3072 + n], x1 = x[b*3072 + 1024 + n], x2 = x[b*3072 + 2048 + n];
  float h[64];
#pragma unroll
  for (int o = 0; o < 64; ++o)
    h[o] = lrelu(W1[o*3]*x0 + W1[o*3+1]*x1 + W1[o*3+2]*x2 + B1[o]);
  _Float16* dst = h2f16 + (size_t)p*128;
  for (int o2 = 0; o2 < 128; ++o2){
    float s = B2[o2];
    const float* wr = &W2[o2*64];
#pragma unroll 8
    for (int k = 0; k < 64; ++k) s += wr[k]*h[k];
    dst[o2] = (_Float16)lrelu(s);
  }
}

// ---------------- STN conv3 (128->1024) WMMA + lrelu + max over N -----------
__global__ void stn_conv3_max(const _Float16* __restrict__ h2f16,
                              const _Float16* __restrict__ sw3h,
                              const float* __restrict__ sb3,
                              float* __restrict__ h3max){
  int b = blockIdx.x, grp = blockIdx.y;              // 16 x 16
  int wv = threadIdx.x >> 5, lane = threadIdx.x & 31;
  int n0 = (grp*4 + wv)*16;                          // output-channel tile base
  const _Float16* A = h2f16 + (size_t)b*131072;      // [1024 pts][128]
  v16h Bf0 = load_b_frag(sw3h, n0, 0);
  v16h Bf1 = load_b_frag(sw3h, n0, 32);
  v16h Bf2 = load_b_frag(sw3h, n0, 64);
  v16h Bf3 = load_b_frag(sw3h, n0, 96);
  float bias = sb3[n0 + (lane & 15)];
  float vmax = -3.4e38f;
  for (int mt = 0; mt < 64; ++mt){
    const _Float16* Ar = A + mt*2048;
    v8f acc = {};
    acc = wmma_f16(load_a_frag(Ar, 0),  Bf0, acc);
    acc = wmma_f16(load_a_frag(Ar, 32), Bf1, acc);
    acc = wmma_f16(load_a_frag(Ar, 64), Bf2, acc);
    acc = wmma_f16(load_a_frag(Ar, 96), Bf3, acc);
#pragma unroll
    for (int r = 0; r < 8; ++r) vmax = fmaxf(vmax, lrelu(acc[r] + bias));
  }
  vmax = fmaxf(vmax, __shfl_xor(vmax, 16, 32));      // combine M-halves (lane, lane+16)
  if (lane < 16) h3max[b*1024 + n0 + lane] = vmax;
}

// ---------------- FC head 1024->256->128->9 (+identity) ---------------------
__global__ void fc_head(const float* __restrict__ h3max,
                        const float* __restrict__ fw1, const float* __restrict__ fb1,
                        const float* __restrict__ fw2, const float* __restrict__ fb2,
                        const float* __restrict__ fw3, const float* __restrict__ fb3,
                        float* __restrict__ transm){
  __shared__ float h[1024], a1[256], a2[128];
  int t = threadIdx.x;
  for (int b = 0; b < 16; ++b){
    for (int i = t; i < 1024; i += 256) h[i] = h3max[b*1024 + i];
    __syncthreads();
    {
      float s = fb1[t];
      const float* wr = fw1 + (size_t)t*1024;
      for (int k = 0; k < 1024; ++k) s += wr[k]*h[k];
      a1[t] = lrelu(s);
    }
    __syncthreads();
    if (t < 128){
      float s = fb2[t];
      const float* wr = fw2 + (size_t)t*256;
      for (int k = 0; k < 256; ++k) s += wr[k]*a1[k];
      a2[t] = lrelu(s);
    }
    __syncthreads();
    if (t < 9){
      float s = fb3[t];
      const float* wr = fw3 + (size_t)t*128;
      for (int k = 0; k < 128; ++k) s += wr[k]*a2[k];
      if ((t & 3) == 0) s += 1.f;                    // eye(3).reshape(9): 0,4,8
      transm[b*9 + t] = s;
    }
    __syncthreads();
  }
}

// ---------------- transform + conv c1 (3->64) + BN1 stats -------------------
__global__ void transform_conv1(const float* __restrict__ x, const float* __restrict__ transm,
                                const float* __restrict__ cw1, const float* __restrict__ cb1,
                                float* __restrict__ c1_pre,
                                float* __restrict__ s1sum, float* __restrict__ s1ssq){
  __shared__ float W1[192], B1[64], ssum[64], sssq[64];
  int tid = threadIdx.x;
  if (tid < 192) W1[tid] = cw1[tid];
  if (tid < 64){ B1[tid] = cb1[tid]; ssum[tid] = 0.f; sssq[tid] = 0.f; }
  __syncthreads();
  int p = blockIdx.x*256 + tid;
  int b = p >> 10, n = p & 1023;
  float x0 = x[b*3072 + n], x1 = x[b*3072 + 1024 + n], x2 = x[b*3072 + 2048 + n];
  const float* T = transm + b*9;                     // y[d] = sum_c x[c]*T[c*3+d]
  float y0 = x0*T[0] + x1*T[3] + x2*T[6];
  float y1 = x0*T[1] + x1*T[4] + x2*T[7];
  float y2 = x0*T[2] + x1*T[5] + x2*T[8];
  float* dst = c1_pre + (size_t)p*64;
#pragma unroll
  for (int o = 0; o < 64; ++o){
    float v = W1[o*3]*y0 + W1[o*3+1]*y1 + W1[o*3+2]*y2 + B1[o];
    dst[o] = v;
    atomicAdd(&ssum[o], v);
    atomicAdd(&sssq[o], v*v);
  }
  __syncthreads();
  if (tid < 64){ atomicAdd(&s1sum[tid], ssum[tid]); atomicAdd(&s1ssq[tid], sssq[tid]); }
}

// ---------------- bn1+lrelu + conv c2 (64->128) + BN2 stats -----------------
__global__ void conv2_main(const float* __restrict__ c1_pre,
                           const float* __restrict__ s1sum, const float* __restrict__ s1ssq,
                           const float* __restrict__ g1, const float* __restrict__ be1,
                           const float* __restrict__ cw2, const float* __restrict__ cb2,
                           float* __restrict__ c2_pre,
                           float* __restrict__ s2sum, float* __restrict__ s2ssq){
  __shared__ float W[128*64];
  __shared__ float scl[64], sft[64], B2[128], ssum[128], sssq[128];
  int tid = threadIdx.x;
  for (int i = tid; i < 8192; i += 256) W[i] = cw2[i];
  if (tid < 64){
    float mean = s1sum[tid]*(1.f/16384.f);
    float var  = s1ssq[tid]*(1.f/16384.f) - mean*mean;
    float rs   = rsqrtf(var + 1e-5f);
    scl[tid] = g1[tid]*rs;
    sft[tid] = be1[tid] - mean*g1[tid]*rs;
  }
  if (tid < 128){ B2[tid] = cb2[tid]; ssum[tid] = 0.f; sssq[tid] = 0.f; }
  __syncthreads();
  int p = blockIdx.x*256 + tid;
  const float* src = c1_pre + (size_t)p*64;
  float a[64];
#pragma unroll
  for (int o = 0; o < 64; ++o) a[o] = lrelu(src[o]*scl[o] + sft[o]);
  float* dst = c2_pre + (size_t)p*128;
  for (int o2 = 0; o2 < 128; ++o2){
    float s = B2[o2];
    const float* wr = &W[o2*64];
#pragma unroll 8
    for (int k = 0; k < 64; ++k) s += wr[k]*a[k];
    dst[o2] = s;
    atomicAdd(&ssum[o2], s);
    atomicAdd(&sssq[o2], s*s);
  }
  __syncthreads();
  if (tid < 128){ atomicAdd(&s2sum[tid], ssum[tid]); atomicAdd(&s2ssq[tid], sssq[tid]); }
}

// ---------------- bn2+lrelu -> f16 A-matrix ---------------------------------
__global__ void bn2_tof16(const float* __restrict__ c2_pre,
                          const float* __restrict__ s2sum, const float* __restrict__ s2ssq,
                          const float* __restrict__ g2, const float* __restrict__ be2,
                          _Float16* __restrict__ a2f16){
  int idx = blockIdx.x*256 + threadIdx.x;            // 2,097,152 elems
  int k = idx & 127;
  float mean = s2sum[k]*(1.f/16384.f);
  float var  = s2ssq[k]*(1.f/16384.f) - mean*mean;
  float rs   = rsqrtf(var + 1e-5f);
  float v = (c2_pre[idx] - mean)*rs*g2[k] + be2[k];
  a2f16[idx] = (_Float16)lrelu(v);
}

// ---------------- main conv3 (128->even 512 ch) WMMA + BN3 stats ------------
__global__ void main_conv3(const _Float16* __restrict__ a2f16,
                           const _Float16* __restrict__ cw3eh, const float* __restrict__ cb3,
                           float* __restrict__ c3even,
                           float* __restrict__ s3sum, float* __restrict__ s3ssq){
  int b = blockIdx.x, grp = blockIdx.y;              // 16 x 8, 4 waves -> 32 j-tiles
  int wv = threadIdx.x >> 5, lane = threadIdx.x & 31;
  int j0 = (grp*4 + wv)*16;                          // even-channel index tile
  const _Float16* A = a2f16 + (size_t)b*131072;
  v16h Bf0 = load_b_frag(cw3eh, j0, 0);
  v16h Bf1 = load_b_frag(cw3eh, j0, 32);
  v16h Bf2 = load_b_frag(cw3eh, j0, 64);
  v16h Bf3 = load_b_frag(cw3eh, j0, 96);
  int jl = lane & 15, hf = lane >> 4;
  float bias = cb3[2*(j0 + jl)];
  float* dst = c3even + (size_t)b*262144 + (j0 + jl);   // [b][w][512]
  float psum = 0.f, pssq = 0.f;
  for (int mt = 0; mt < 64; ++mt){
    const _Float16* Ar = A + mt*2048;
    v8f acc = {};
    acc = wmma_f16(load_a_frag(Ar, 0),  Bf0, acc);
    acc = wmma_f16(load_a_frag(Ar, 32), Bf1, acc);
    acc = wmma_f16(load_a_frag(Ar, 64), Bf2, acc);
    acc = wmma_f16(load_a_frag(Ar, 96), Bf3, acc);
#pragma unroll
    for (int r = 0; r < 8; ++r){
      float v = acc[r] + bias;
      psum += v; pssq += v*v;                        // stats need ALL n
      if ((r & 1) == 0){                             // even m -> even global n kept
        int wIdx = mt*8 + (r >> 1) + hf*4;           // w = (mt*16 + m)/2
        dst[(size_t)wIdx*512] = v;
      }
    }
  }
  psum += __shfl_xor(psum, 16, 32);
  pssq += __shfl_xor(pssq, 16, 32);
  if (lane < 16){
    atomicAdd(&s3sum[j0 + lane], psum);
    atomicAdd(&s3ssq[j0 + lane], pssq);
  }
}

// ---------------- bn3+lrelu + top-32-of-512 descending per (b,w) ------------
__global__ void bn3_top32(const float* __restrict__ c3even,
                          const float* __restrict__ s3sum, const float* __restrict__ s3ssq,
                          const float* __restrict__ g3, const float* __restrict__ be3,
                          float* __restrict__ tmat){
  int wv = threadIdx.x >> 5, lane = threadIdx.x & 31;
  int gw = blockIdx.x*8 + wv;                        // 8192 waves
  int b = gw >> 9, w = gw & 511;
  const float* src = c3even + ((size_t)b*512 + w)*512;
  float vals[16];
  const float inv = 1.f/16384.f;
#pragma unroll
  for (int q = 0; q < 16; ++q){
    int j = q*32 + lane;
    float mean = s3sum[j]*inv;
    float var  = s3ssq[j]*inv - mean*mean;
    float rs   = rsqrtf(var + 1e-5f);
    float v = (src[j] - mean)*rs*g3[2*j] + be3[2*j];
    vals[q] = lrelu(v);
  }
  const float NEG = -3.4e38f;
  for (int r = 0; r < 32; ++r){
    float lmax = NEG; int lidx = 0;
#pragma unroll
    for (int q = 0; q < 16; ++q){ if (vals[q] > lmax){ lmax = vals[q]; lidx = q; } }
    float wmax = lmax;
#pragma unroll
    for (int off = 16; off >= 1; off >>= 1) wmax = fmaxf(wmax, __shfl_xor(wmax, off, 32));
    unsigned m = (unsigned)__ballot(lmax == wmax);
    int winner = __ffs(m) - 1;                       // lowest lane holding the max
    if (lane == winner){
#pragma unroll
      for (int q = 0; q < 16; ++q) if (q == lidx) vals[q] = NEG;
    }
    if (lane == 0) tmat[((size_t)(b*32 + r))*512 + w] = wmax;
  }
}

// ---------------- lw contraction (1.07 GB stream) + affine max --------------
__global__ void lc_max(const float* __restrict__ tmat, const float* __restrict__ lw,
                       const float* __restrict__ dw, const float* __restrict__ db,
                       float* __restrict__ out){
  int o = blockIdx.x;                                // 1024
  int tid = threadIdx.x;                             // 128 threads x 4 w
  int w0 = tid*4;
  const float* L = lw + (size_t)o*262144;
  f4 acc = {0.f, 0.f, 0.f, 0.f};
#pragma unroll 4
  for (int i = 0; i < 512; ++i){
    f4 lv = __builtin_nontemporal_load((const f4*)(L + (size_t)i*512 + w0));
    f4 tv = *(const f4*)(tmat + (size_t)i*512 + w0);
    acc += lv*tv;
  }
  float mx = fmaxf(fmaxf(acc[0], acc[1]), fmaxf(acc[2], acc[3]));
  float mn = fminf(fminf(acc[0], acc[1]), fminf(acc[2], acc[3]));
  __shared__ float smx[128], smn[128];
  smx[tid] = mx; smn[tid] = mn;
  __syncthreads();
  for (int s = 64; s > 0; s >>= 1){
    if (tid < s){ smx[tid] = fmaxf(smx[tid], smx[tid+s]); smn[tid] = fminf(smn[tid], smn[tid+s]); }
    __syncthreads();
  }
  if (tid < 16){
    float d = dw[tid];
    out[tid*1024 + o] = (d >= 0.f ? d*smx[0] : d*smn[0]) + db[tid];
  }
}

// ---------------- host ------------------------------------------------------
extern "C" void kernel_launch(void* const* d_in, const int* in_sizes, int n_in,
                              void* d_out, int out_size, void* d_ws, size_t ws_size,
                              hipStream_t stream){
  const float* x   = (const float*)d_in[0];
  const float* sw1 = (const float*)d_in[1];
  const float* sb1 = (const float*)d_in[2];
  const float* sw2 = (const float*)d_in[3];
  const float* sb2 = (const float*)d_in[4];
  const float* sw3 = (const float*)d_in[5];
  const float* sb3 = (const float*)d_in[6];
  const float* fw1 = (const float*)d_in[7];
  const float* fb1 = (const float*)d_in[8];
  const float* fw2 = (const float*)d_in[9];
  const float* fb2 = (const float*)d_in[10];
  const float* fw3 = (const float*)d_in[11];
  const float* fb3 = (const float*)d_in[12];
  const float* cw1 = (const float*)d_in[13];
  const float* cb1 = (const float*)d_in[14];
  const float* cw2 = (const float*)d_in[15];
  const float* cb2 = (const float*)d_in[16];
  const float* cw3 = (const float*)d_in[17];
  const float* cb3 = (const float*)d_in[18];
  const float* g1  = (const float*)d_in[19];
  const float* be1 = (const float*)d_in[20];
  const float* g2  = (const float*)d_in[21];
  const float* be2 = (const float*)d_in[22];
  const float* g3  = (const float*)d_in[23];
  const float* be3 = (const float*)d_in[24];
  const float* lw  = (const float*)d_in[25];
  const float* dw  = (const float*)d_in[26];
  const float* db  = (const float*)d_in[27];
  float* out = (float*)d_out;

  char* ws = (char*)d_ws;
  size_t off = 0;
  auto alloc = [&](size_t bytes)->char*{
    char* p = ws + off; off += (bytes + 255) & ~(size_t)255; return p;
  };
  _Float16* h2f16 = (_Float16*)alloc((size_t)16384*128*2); // 4 MB
  _Float16* a2f16 = (_Float16*)alloc((size_t)16384*128*2); // 4 MB
  _Float16* sw3h  = (_Float16*)alloc((size_t)131072*2);
  _Float16* cw3eh = (_Float16*)alloc((size_t)65536*2);
  float* h3max  = (float*)alloc((size_t)16384*4);
  float* transm = (float*)alloc((size_t)144*4);
  float* c1_pre = (float*)alloc((size_t)16384*64*4);       // 4 MB
  float* c2_pre = (float*)alloc((size_t)16384*128*4);      // 8 MB
  float* c3even = (float*)alloc((size_t)16*512*512*4);     // 16 MB
  float* tmat   = (float*)alloc((size_t)512*512*4);        // 1 MB
  float* stats  = (float*)alloc((size_t)1408*4);
  float* s1sum = stats,       *s1ssq = stats + 64;
  float* s2sum = stats + 128, *s2ssq = stats + 256;
  float* s3sum = stats + 384, *s3ssq = stats + 896;

  init_convert<<<512, 256, 0, stream>>>(sw3, cw3, sw3h, cw3eh, stats);
  stn_conv12<<<64, 256, 0, stream>>>(x, sw1, sb1, sw2, sb2, h2f16);
  stn_conv3_max<<<dim3(16,16), 128, 0, stream>>>(h2f16, sw3h, sb3, h3max);
  fc_head<<<1, 256, 0, stream>>>(h3max, fw1, fb1, fw2, fb2, fw3, fb3, transm);
  transform_conv1<<<64, 256, 0, stream>>>(x, transm, cw1, cb1, c1_pre, s1sum, s1ssq);
  conv2_main<<<64, 256, 0, stream>>>(c1_pre, s1sum, s1ssq, g1, be1, cw2, cb2,
                                     c2_pre, s2sum, s2ssq);
  bn2_tof16<<<8192, 256, 0, stream>>>(c2_pre, s2sum, s2ssq, g2, be2, a2f16);
  main_conv3<<<dim3(16,8), 128, 0, stream>>>(a2f16, cw3eh, cb3, c3even, s3sum, s3ssq);
  bn3_top32<<<1024, 256, 0, stream>>>(c3even, s3sum, s3ssq, g3, be3, tmat);
  lc_max<<<1024, 128, 0, stream>>>(tmat, lw, dw, db, out);
}
